// LSTMbloodPressure_86852828660080
// MI455X (gfx1250) — matrix operand order repaired and checked by reading, not compiled, not run on hardware
//
#include <hip/hip_runtime.h>
#include <hip/hip_bf16.h>
#include <math.h>

typedef __attribute__((ext_vector_type(16))) _Float16 v16h;
typedef __attribute__((ext_vector_type(8)))  float    v8f;
typedef __attribute__((ext_vector_type(4)))  float    v4f;   // native vector for NT loads

#define T_STEPS 256
#define BATCH   1000
#define INSZ    300
#define HID     2
#define G4      8                       // 4*HIDDEN
#define M_ROWS  (T_STEPS * BATCH)       // 256000
#define NTILES  (M_ROWS / 16)           // 16000 full 16-row tiles
#define KCHUNKS 10                      // ceil(300/32), K padded to 320

__device__ __forceinline__ float sigmoidf_(float x) {
    return 1.0f / (1.0f + __expf(-x));
}

// ---------------------------------------------------------------------------
// Kernel 1: xg0[r][g] = sum_k x[r][k] * W_ih0[g][k] + (b_ih0[g] + b_hh0[g])
// r = t*BATCH + b (x is [T,B,300] row-major). GEMM M=256000, N=8(pad 16),
// K=300(pad 320) via V_WMMA_F32_16X16X32_F16.
// Memory-bound: 307 MB of x streamed once -> non-temporal loads; one 16-row
// tile per wave (16000 waves) for max memory-level parallelism. W fragments
// are built once per block into LDS and re-read by all 8 waves.
// ---------------------------------------------------------------------------
__global__ void __launch_bounds__(256)
xg0_wmma_kernel(const float* __restrict__ x,
                const float* __restrict__ W_ih0,
                const float* __restrict__ b_ih0,
                const float* __restrict__ b_hh0,
                float* __restrict__ xg0)
{
    const int lane = threadIdx.x & 31;
    const int half = lane >> 4;     // lane group 0..15 vs 16..31
    const int nm   = lane & 15;     // M for A frag, N for B/D frags

    // B fragments (W_ih0 transposed to [K x N], f16, zero-padded) are the
    // same for every wave: wave 0 builds them into LDS, everyone loads back.
    // B layout: lane holds column N = nm; element j holds K = kc*32+8*half+j (j<8),
    // kc*32+16+8*half+(j-8) (j>=8).
    __shared__ v16h bshared[KCHUNKS][32];
    if (threadIdx.x < 32) {
#pragma unroll
        for (int kc = 0; kc < KCHUNKS; ++kc) {
            v16h f;
#pragma unroll
            for (int j = 0; j < 16; ++j) {
                const int k = kc * 32 + ((j < 8) ? half * 8 + j : 16 + half * 8 + (j - 8));
                const float w = (nm < G4 && k < INSZ) ? W_ih0[nm * INSZ + k] : 0.0f;
                f[j] = (_Float16)w;
            }
            bshared[kc][lane] = f;
        }
    }
    __syncthreads();

    v16h bfrag[KCHUNKS];
#pragma unroll
    for (int kc = 0; kc < KCHUNKS; ++kc) bfrag[kc] = bshared[kc][lane];

    const float bias = (nm < G4) ? (b_ih0[nm] + b_hh0[nm]) : 0.0f;

    // Exactly one tile per wave: 2000 blocks * 8 waves = 16000 waves.
    const int tile = (blockIdx.x * blockDim.x + threadIdx.x) >> 5;
    if (tile >= NTILES) return;   // wave-uniform; full grid covers exactly

    const float* rowp = x + (size_t)(tile * 16 + nm) * INSZ;  // A row M=nm
    v8f acc = {};

#pragma unroll
    for (int kc = 0; kc < KCHUNKS; ++kc) {
        // A layout: lane holds row M=nm; elems 0..7 = K[k0+8*half .. +7],
        // elems 8..15 = K[k0+16+8*half .. +7].
        v16h a;
        const int k0 = kc * 32;
        if (kc < KCHUNKS - 1) {
            // 16B-aligned (row base r*1200B, offsets mult. of 32B), streamed
            // once -> non-temporal so the 307MB stream doesn't rinse L2.
            const v4f* p1 = reinterpret_cast<const v4f*>(rowp + k0 + half * 8);
            const v4f* p2 = reinterpret_cast<const v4f*>(rowp + k0 + 16 + half * 8);
            const v4f u0 = __builtin_nontemporal_load(p1);
            const v4f u1 = __builtin_nontemporal_load(p1 + 1);
            const v4f u2 = __builtin_nontemporal_load(p2);
            const v4f u3 = __builtin_nontemporal_load(p2 + 1);
#pragma unroll
            for (int j = 0; j < 4; ++j) {
                a[j]      = (_Float16)u0[j];
                a[4 + j]  = (_Float16)u1[j];
                a[8 + j]  = (_Float16)u2[j];
                a[12 + j] = (_Float16)u3[j];
            }
        } else {
            // K tail 288..319: guard k<300, avoid OOB past x allocation.
#pragma unroll
            for (int j = 0; j < 8; ++j) {
                const int k = k0 + half * 8 + j;
                a[j] = (_Float16)((k < INSZ) ? rowp[k] : 0.0f);
            }
#pragma unroll
            for (int j = 8; j < 16; ++j) a[j] = (_Float16)0.0f;  // K >= 304
        }
        acc = __builtin_amdgcn_wmma_f32_16x16x32_f16(
            /*neg_a=*/false, a, /*neg_b=*/false, bfrag[kc],
            /*c_mod=*/(short)0, acc, /*reuse_a=*/false, /*reuse_b=*/false);
    }

    // D layout: lane (half,nm), VGPR v -> D[M = 8*half + v][N = nm].
    if (nm < G4) {
#pragma unroll
        for (int v = 0; v < 8; ++v) {
            const int r = tile * 16 + half * 8 + v;
            xg0[(size_t)r * G4 + nm] = acc[v] + bias;
        }
    }
}

// ---------------------------------------------------------------------------
// Kernel 2: fused 2-layer LSTM recurrence + final linear, one thread per
// batch element. All weight matrices are 8x2 -> register resident. Reads of
// xg0 are fully coalesced (32B per lane, contiguous across lanes).
// ---------------------------------------------------------------------------
__global__ void __launch_bounds__(256)
lstm_rec_kernel(const float* __restrict__ xg0,
                const float* __restrict__ h0_init,
                const float* __restrict__ c0_init,
                const float* __restrict__ W_hh0,
                const float* __restrict__ W_ih1,
                const float* __restrict__ W_hh1,
                const float* __restrict__ b_ih1,
                const float* __restrict__ b_hh1,
                const float* __restrict__ W_lin,
                const float* __restrict__ b_lin,
                float* __restrict__ out)
{
    const int b = blockIdx.x * blockDim.x + threadIdx.x;
    if (b >= BATCH) return;

    float whh0[G4][HID], wih1[G4][HID], whh1[G4][HID], b1[G4];
#pragma unroll
    for (int g = 0; g < G4; ++g) {
        whh0[g][0] = W_hh0[g * HID + 0]; whh0[g][1] = W_hh0[g * HID + 1];
        wih1[g][0] = W_ih1[g * HID + 0]; wih1[g][1] = W_ih1[g * HID + 1];
        whh1[g][0] = W_hh1[g * HID + 0]; whh1[g][1] = W_hh1[g * HID + 1];
        b1[g] = b_ih1[g] + b_hh1[g];
    }
    const float wl0 = W_lin[0], wl1 = W_lin[1], bl = b_lin[0];

    // h0/c0 are [LAYERS, B, HID]
    float h00 = h0_init[b * HID + 0],               h01 = h0_init[b * HID + 1];
    float c00 = c0_init[b * HID + 0],               c01 = c0_init[b * HID + 1];
    float h10 = h0_init[BATCH * HID + b * HID + 0], h11 = h0_init[BATCH * HID + b * HID + 1];
    float c10 = c0_init[BATCH * HID + b * HID + 0], c11 = c0_init[BATCH * HID + b * HID + 1];

    for (int t = 0; t < T_STEPS; ++t) {
        const float* xg = xg0 + ((size_t)t * BATCH + b) * G4;

        // layer 0 gates (input projection + biases already in xg)
        float g0[G4];
#pragma unroll
        for (int g = 0; g < G4; ++g)
            g0[g] = xg[g] + whh0[g][0] * h00 + whh0[g][1] * h01;

        {   // gate order i, f, g, o (PyTorch), 2 units each
            const float i0 = sigmoidf_(g0[0]), i1 = sigmoidf_(g0[1]);
            const float f0 = sigmoidf_(g0[2]), f1 = sigmoidf_(g0[3]);
            const float t0 = tanhf(g0[4]),     t1 = tanhf(g0[5]);
            const float o0 = sigmoidf_(g0[6]), o1 = sigmoidf_(g0[7]);
            c00 = f0 * c00 + i0 * t0;  c01 = f1 * c01 + i1 * t1;
            h00 = o0 * tanhf(c00);     h01 = o1 * tanhf(c01);
        }

        // layer 1 gates
        float g1[G4];
#pragma unroll
        for (int g = 0; g < G4; ++g)
            g1[g] = b1[g] + wih1[g][0] * h00 + wih1[g][1] * h01
                          + whh1[g][0] * h10 + whh1[g][1] * h11;

        {
            const float i0 = sigmoidf_(g1[0]), i1 = sigmoidf_(g1[1]);
            const float f0 = sigmoidf_(g1[2]), f1 = sigmoidf_(g1[3]);
            const float t0 = tanhf(g1[4]),     t1 = tanhf(g1[5]);
            const float o0 = sigmoidf_(g1[6]), o1 = sigmoidf_(g1[7]);
            c10 = f0 * c10 + i0 * t0;  c11 = f1 * c11 + i1 * t1;
            h10 = o0 * tanhf(c10);     h11 = o1 * tanhf(c11);
        }
    }

    out[b] = wl0 * h10 + wl1 * h11 + bl;
}

// ---------------------------------------------------------------------------
extern "C" void kernel_launch(void* const* d_in, const int* in_sizes, int n_in,
                              void* d_out, int out_size, void* d_ws, size_t ws_size,
                              hipStream_t stream)
{
    (void)in_sizes; (void)n_in; (void)out_size; (void)ws_size;

    const float* x     = (const float*)d_in[0];
    const float* h0    = (const float*)d_in[1];
    const float* c0    = (const float*)d_in[2];
    const float* W_ih0 = (const float*)d_in[3];
    const float* W_hh0 = (const float*)d_in[4];
    const float* b_ih0 = (const float*)d_in[5];
    const float* b_hh0 = (const float*)d_in[6];
    const float* W_ih1 = (const float*)d_in[7];
    const float* W_hh1 = (const float*)d_in[8];
    const float* b_ih1 = (const float*)d_in[9];
    const float* b_hh1 = (const float*)d_in[10];
    const float* W_lin = (const float*)d_in[11];
    const float* b_lin = (const float*)d_in[12];
    float* out = (float*)d_out;

    float* xg0 = (float*)d_ws;  // [T*B][8] f32 = 8.192 MB scratch

    // One 16-row tile per wave: 2000 blocks * 8 waves = 16000 waves.
    xg0_wmma_kernel<<<2000, 256, 0, stream>>>(x, W_ih0, b_ih0, b_hh0, xg0);

    lstm_rec_kernel<<<(BATCH + 255) / 256, 256, 0, stream>>>(
        xg0, h0, c0, W_hh0, W_ih1, W_hh1, b_ih1, b_hh1, W_lin, b_lin, out);
}
